// RNN_Generator_5755256177029
// MI455X (gfx1250) — matrix operand order repaired and checked by reading, compiled, not run on hardware
//
#include <hip/hip_runtime.h>
#include <hip/hip_bf16.h>

// ---------------- problem constants ----------------
constexpr int En = 256;          // embedding dim
constexpr int Dn = 512;          // hidden dim
constexpr int Kn = 16;           // neighbors
constexpr int Bn = 128;          // batch
constexpr int Sn = 128;          // sequence
constexpr float BETAc = 0.1f;
constexpr float MAX_TIMEc = 1000.0f;
constexpr int Lc = 1 + (Sn - 1) * (Kn - 1);   // 1906 candidate slots
constexpr int Rc = 1 + (Sn - 1) * Kn;         // 2033 nrec slots
constexpr int NTHREADS = 1024;                // 32 wave32 waves
constexpr int NWAVES = 32;
constexpr int BS = Bn * Sn;

typedef __attribute__((ext_vector_type(16))) __bf16 v16bf;
typedef __attribute__((ext_vector_type(8)))  __bf16 v8bf;
typedef __attribute__((ext_vector_type(8)))  float  v8f;

__device__ __forceinline__ v16bf cat16(v8bf lo, v8bf hi) {
    return __builtin_shufflevector(lo, hi, 0, 1, 2, 3, 4, 5, 6, 7,
                                           8, 9, 10, 11, 12, 13, 14, 15);
}

// A-fragment load: lane owns row (rowTile*16 + lane&15), two contiguous 128-bit runs.
__device__ __forceinline__ v16bf ldA(const __bf16* __restrict__ S, int rowTile,
                                     int kb, int stride, int koffA, int lane) {
    const int arow = rowTile * 16 + (lane & 15);
    return cat16(*(const v8bf*)&S[arow * stride + kb + koffA],
                 *(const v8bf*)&S[arow * stride + kb + 8 + koffA]);
}

__device__ __forceinline__ unsigned packbf(float a, float b) {
    __bf16 x = (__bf16)a, y = (__bf16)b;
    unsigned short ux, uy;
    __builtin_memcpy(&ux, &x, 2);
    __builtin_memcpy(&uy, &y, 2);
    return ((unsigned)uy << 16) | (unsigned)ux;
}

// ---------------- deterministic per-(t,b) uniform ----------------
__device__ __forceinline__ float rng01(unsigned t, unsigned b) {
    unsigned x = t * 0x9E3779B9u ^ (b * 0x85EBCA6Bu + 0xC2B2AE35u);
    x ^= x >> 16; x *= 0x7FEB352Du;
    x ^= x >> 15; x *= 0x846CA68Bu;
    x ^= x >> 16;
    return (float)(x >> 8) * (1.0f / 16777216.0f);
}

// ---------------- weight fp32 -> bf16 TRANSPOSED prep ----------------
__global__ void prep_weights(const float* __restrict__ Wel,
                             const float* __restrict__ Wih,
                             const float* __restrict__ Whh,
                             __bf16* __restrict__ bWelT,
                             __bf16* __restrict__ bWihT,
                             __bf16* __restrict__ bWhhT) {
    int i = blockIdx.x * blockDim.x + threadIdx.x;
    if (i < En * Dn) {
        int k = i / Dn, n = i % Dn;
        bWelT[n * En + k] = (__bf16)Wel[i];
    }
    if (i < Dn * Dn) {
        int k = i / Dn, n = i % Dn;
        bWihT[n * Dn + k] = (__bf16)Wih[i];
        bWhhT[n * Dn + k] = (__bf16)Whh[i];
    }
}

// 16 packed bf16 pairs (one 4-tile group of D results), kept in VGPRs.
struct Pk16 { unsigned v[16]; };

// ---- phase-2 group: x = leaky(vec @ W_el + b_el) for row tiles [4G, 4G+3] ----
template <int G>
__device__ __forceinline__ void el_group(const __bf16* __restrict__ sVec,
                                         __bf16* __restrict__ sX,
                                         const __bf16* __restrict__ bWelT,
                                         float bias, int lane, int col,
                                         int koffA, int koffB, int crbase) {
    v8f acc[4];
#pragma unroll
    for (int rt = 0; rt < 4; ++rt)
#pragma unroll
        for (int v = 0; v < 8; ++v) acc[rt][v] = bias;

#pragma unroll 1
    for (int kb = 0; kb < En; kb += 32) {
        const v8bf* pB = (const v8bf*)&bWelT[col * En + kb + koffB];
        v16bf bm = cat16(pB[0], pB[1]);
        v16bf a0 = ldA(sVec, G * 4 + 0, kb, En, koffA, lane);
        v16bf a1 = ldA(sVec, G * 4 + 1, kb, En, koffA, lane);
        v16bf a2 = ldA(sVec, G * 4 + 2, kb, En, koffA, lane);
        v16bf a3 = ldA(sVec, G * 4 + 3, kb, En, koffA, lane);
        acc[0] = __builtin_amdgcn_wmma_f32_16x16x32_bf16(false, a0, false, bm, (short)0, acc[0], false, false);
        acc[1] = __builtin_amdgcn_wmma_f32_16x16x32_bf16(false, a1, false, bm, (short)0, acc[1], false, false);
        acc[2] = __builtin_amdgcn_wmma_f32_16x16x32_bf16(false, a2, false, bm, (short)0, acc[2], false, false);
        acc[3] = __builtin_amdgcn_wmma_f32_16x16x32_bf16(false, a3, false, bm, (short)0, acc[3], false, false);
    }
    // sX is not read until after the next barrier -> store immediately
#pragma unroll
    for (int rt = 0; rt < 4; ++rt) {
        const int rb = (G * 4 + rt) * 16 + crbase;
#pragma unroll
        for (int v = 0; v < 8; ++v) {
            float f = acc[rt][v];
            f = (f >= 0.0f) ? f : 0.01f * f;
            sX[(rb + v) * Dn + col] = (__bf16)f;
        }
    }
}

// ---- phase-3 group: tanh(x@W_ih + h@W_hh + bias) for row tiles [4G, 4G+3],
//      result returned packed (bf16 pairs) so old sH stays readable until the barrier.
template <int G>
__device__ __forceinline__ Pk16 hh_group(const __bf16* __restrict__ sX,
                                         const __bf16* __restrict__ sH,
                                         const __bf16* __restrict__ bWihT,
                                         const __bf16* __restrict__ bWhhT,
                                         float bias, int lane, int col,
                                         int koffA, int koffB) {
    v8f acc[4];
#pragma unroll
    for (int rt = 0; rt < 4; ++rt)
#pragma unroll
        for (int v = 0; v < 8; ++v) acc[rt][v] = bias;

#pragma unroll 1
    for (int kb = 0; kb < Dn; kb += 32) {
        const v8bf* pBi = (const v8bf*)&bWihT[col * Dn + kb + koffB];
        v16bf bih = cat16(pBi[0], pBi[1]);
        v16bf a0 = ldA(sX, G * 4 + 0, kb, Dn, koffA, lane);
        v16bf a1 = ldA(sX, G * 4 + 1, kb, Dn, koffA, lane);
        v16bf a2 = ldA(sX, G * 4 + 2, kb, Dn, koffA, lane);
        v16bf a3 = ldA(sX, G * 4 + 3, kb, Dn, koffA, lane);
        acc[0] = __builtin_amdgcn_wmma_f32_16x16x32_bf16(false, a0, false, bih, (short)0, acc[0], false, false);
        acc[1] = __builtin_amdgcn_wmma_f32_16x16x32_bf16(false, a1, false, bih, (short)0, acc[1], false, false);
        acc[2] = __builtin_amdgcn_wmma_f32_16x16x32_bf16(false, a2, false, bih, (short)0, acc[2], false, false);
        acc[3] = __builtin_amdgcn_wmma_f32_16x16x32_bf16(false, a3, false, bih, (short)0, acc[3], false, false);
    }
#pragma unroll 1
    for (int kb = 0; kb < Dn; kb += 32) {
        const v8bf* pBh = (const v8bf*)&bWhhT[col * Dn + kb + koffB];
        v16bf bhh = cat16(pBh[0], pBh[1]);
        v16bf a0 = ldA(sH, G * 4 + 0, kb, Dn, koffA, lane);
        v16bf a1 = ldA(sH, G * 4 + 1, kb, Dn, koffA, lane);
        v16bf a2 = ldA(sH, G * 4 + 2, kb, Dn, koffA, lane);
        v16bf a3 = ldA(sH, G * 4 + 3, kb, Dn, koffA, lane);
        acc[0] = __builtin_amdgcn_wmma_f32_16x16x32_bf16(false, a0, false, bhh, (short)0, acc[0], false, false);
        acc[1] = __builtin_amdgcn_wmma_f32_16x16x32_bf16(false, a1, false, bhh, (short)0, acc[1], false, false);
        acc[2] = __builtin_amdgcn_wmma_f32_16x16x32_bf16(false, a2, false, bhh, (short)0, acc[2], false, false);
        acc[3] = __builtin_amdgcn_wmma_f32_16x16x32_bf16(false, a3, false, bhh, (short)0, acc[3], false, false);
    }
    Pk16 r;
#pragma unroll
    for (int rt = 0; rt < 4; ++rt)
#pragma unroll
        for (int j = 0; j < 4; ++j)
            r.v[rt * 4 + j] = packbf(tanhf(acc[rt][j * 2]), tanhf(acc[rt][j * 2 + 1]));
    return r;
}

// ---------------- persistent generator: one workgroup runs the scan ----------------
__global__ __launch_bounds__(NTHREADS, 1)
void rnn_generator(const int*   __restrict__ marker,
                   const float* __restrict__ timed,
                   const float* __restrict__ maskd,
                   const float* __restrict__ embedding,
                   const int*   __restrict__ neighbor_list,
                   const float* __restrict__ neighbor_prob,
                   const float* __restrict__ W_te, const float* __restrict__ b_te,
                   const __bf16* __restrict__ bWelT, const float* __restrict__ b_el,
                   const __bf16* __restrict__ bWihT, const float* __restrict__ b_ih,
                   const __bf16* __restrict__ bWhhT, const float* __restrict__ b_hh,
                   const float* __restrict__ W_time, const float* __restrict__ b_time,
                   const float* __restrict__ W_mk,   const float* __restrict__ b_mk,
                   float* __restrict__ out,
                   int*   __restrict__ cand,
                   float* __restrict__ prob,
                   float* __restrict__ nrec,
                   float* __restrict__ lastT,
                   int*   __restrict__ lastM,
                   int*   __restrict__ chosenA,
                   float* __restrict__ newT,
                   float* __restrict__ hdot,
                   float* __restrict__ score,
                   int*   __restrict__ neighB,
                   float* __restrict__ pneighB) {
    extern __shared__ __bf16 smem[];
    __bf16* sVec = smem;                // Bn x En  (64 KB)
    __bf16* sX   = sVec + Bn * En;      // Bn x Dn  (128 KB)
    __bf16* sH   = sX   + Bn * Dn;      // Bn x Dn  (128 KB)  -> 320 KB total

    const int tid  = threadIdx.x;
    const int lane = tid & 31;
    const int wave = tid >> 5;

    if (tid == 0) {
        __builtin_prefetch(bWelT, 0, 3);
        __builtin_prefetch(bWihT, 0, 3);
        __builtin_prefetch(bWhhT, 0, 3);
    }

    // ---------------- init carry / outputs ----------------
    for (int i = tid; i < Bn * Dn; i += NTHREADS) sH[i] = (__bf16)0.0f;
    for (int i = tid; i < Bn * Lc; i += NTHREADS) { cand[i] = 0; prob[i] = 0.0f; }
    for (int i = tid; i < Bn * Rc; i += NTHREADS) nrec[i] = 1.0f;
    if (tid < Bn) {
        int b = tid;
        int m0 = marker[b * Sn];
        cand[b * Lc] = m0;
        prob[b * Lc] = 1.0f;
        lastT[b] = timed[b * Sn];
        lastM[b] = m0;
        chosenA[b] = 0;
        out[0 * BS + b * Sn] = (float)m0;
        out[1 * BS + b * Sn] = timed[b * Sn];
        out[2 * BS + b * Sn] = maskd[b * Sn];
        out[3 * BS + b * Sn] = 1.0f;
        out[4 * BS + b * Sn] = 1.0f;
    }
    __threadfence();
    __syncthreads();

    const int koffA  = (lane < 16) ? 0 : 16;
    const int koffB  = (lane < 16) ? 0 : 16;
    const int col    = wave * 16 + (lane & 15);
    const int crbase = (lane < 16) ? 0 : 8;

    for (int t = 0; t < Sn - 1; ++t) {
        // ---- Phase 1: vec = embedding[last_m] + BETA*(last_t*W_te + b_te) -> LDS bf16 ----
        for (int i = tid; i < Bn * (En / 8); i += NTHREADS) {
            int b  = i / (En / 8);
            int e8 = (i % (En / 8)) * 8;
            float lt = lastT[b];
            const float4* emb = (const float4*)&embedding[(long)lastM[b] * En + e8];
            const float4* wte = (const float4*)&W_te[e8];
            const float4* bte = (const float4*)&b_te[e8];
            float4 v0 = emb[0], v1 = emb[1];
            float4 w0 = wte[0], w1 = wte[1];
            float4 c0 = bte[0], c1 = bte[1];
            v8bf r;
            r[0] = (__bf16)(v0.x + BETAc * (lt * w0.x + c0.x));
            r[1] = (__bf16)(v0.y + BETAc * (lt * w0.y + c0.y));
            r[2] = (__bf16)(v0.z + BETAc * (lt * w0.z + c0.z));
            r[3] = (__bf16)(v0.w + BETAc * (lt * w0.w + c0.w));
            r[4] = (__bf16)(v1.x + BETAc * (lt * w1.x + c1.x));
            r[5] = (__bf16)(v1.y + BETAc * (lt * w1.y + c1.y));
            r[6] = (__bf16)(v1.z + BETAc * (lt * w1.z + c1.z));
            r[7] = (__bf16)(v1.w + BETAc * (lt * w1.w + c1.w));
            *(v8bf*)&sVec[b * En + e8] = r;
        }
        __syncthreads();

        // ---- Phase 2: x = leaky(vec @ W_el + b_el), two 4-tile groups ----
        {
            const float bias = b_el[col];
            el_group<0>(sVec, sX, bWelT, bias, lane, col, koffA, koffB, crbase);
            el_group<1>(sVec, sX, bWelT, bias, lane, col, koffA, koffB, crbase);
        }
        __syncthreads();

        // ---- Phase 3: h = tanh(x @ W_ih + h @ W_hh + b), packed staging + barrier ----
        {
            const float bias = b_ih[col] + b_hh[col];
            Pk16 p0 = hh_group<0>(sX, sH, bWihT, bWhhT, bias, lane, col, koffA, koffB);
            Pk16 p1 = hh_group<1>(sX, sH, bWihT, bWhhT, bias, lane, col, koffA, koffB);
            __syncthreads();   // every wave done READING old sH
            unsigned short* sH16 = (unsigned short*)sH;
#pragma unroll
            for (int rt = 0; rt < 4; ++rt) {
                const int rb0 = rt * 16 + crbase;
                const int rb1 = (4 + rt) * 16 + crbase;
#pragma unroll
                for (int j = 0; j < 4; ++j) {
                    unsigned a = p0.v[rt * 4 + j];
                    unsigned b = p1.v[rt * 4 + j];
                    sH16[(rb0 + 2 * j) * Dn + col]     = (unsigned short)(a & 0xffffu);
                    sH16[(rb0 + 2 * j + 1) * Dn + col] = (unsigned short)(a >> 16);
                    sH16[(rb1 + 2 * j) * Dn + col]     = (unsigned short)(b & 0xffffu);
                    sH16[(rb1 + 2 * j + 1) * Dn + col] = (unsigned short)(b >> 16);
                }
            }
        }
        __syncthreads();

        // ---- Phase 4: new_t = last_t + softplus(h@W_time + b_time); hdot[b] = h.W_mk[256:] ----
        for (int b = wave; b < Bn; b += NWAVES) {
            const v8bf* hp = (const v8bf*)&sH[b * Dn + lane * 16];
            v8bf h0 = hp[0], h1 = hp[1];
            const float4* wt = (const float4*)&W_time[lane * 16];
            const float4* wm = (const float4*)&W_mk[En + lane * 16];
            float s1 = 0.0f, s2 = 0.0f;
#pragma unroll
            for (int q = 0; q < 4; ++q) {
                float4 a = wt[q], c = wm[q];
                float e0 = (q < 2) ? (float)h0[q * 4 + 0] : (float)h1[(q - 2) * 4 + 0];
                float e1 = (q < 2) ? (float)h0[q * 4 + 1] : (float)h1[(q - 2) * 4 + 1];
                float e2 = (q < 2) ? (float)h0[q * 4 + 2] : (float)h1[(q - 2) * 4 + 2];
                float e3 = (q < 2) ? (float)h0[q * 4 + 3] : (float)h1[(q - 2) * 4 + 3];
                s1 += e0 * a.x + e1 * a.y + e2 * a.z + e3 * a.w;
                s2 += e0 * c.x + e1 * c.y + e2 * c.z + e3 * c.w;
            }
#pragma unroll
            for (int off = 16; off > 0; off >>= 1) {
                s1 += __shfl_down(s1, off, 32);
                s2 += __shfl_down(s2, off, 32);
            }
            if (lane == 0) {
                float z  = s1 + b_time[0];
                float sp = (z > 20.0f) ? z : log1pf(expf(z));
                newT[b] = lastT[b] + sp;
                hdot[b] = s2;
            }
        }
        __threadfence();
        __syncthreads();

        // ---- Phase 5: neighbor gather + nrec update ----
        for (int i = tid; i < Bn * Kn; i += NTHREADS) {
            int b = i / Kn, k = i % Kn;
            int lm = lastM[b];
            int nb = neighbor_list[(long)lm * Kn + k];
            float pn = neighbor_prob[(long)lm * Kn + k];
            neighB[i]  = nb;
            pneighB[i] = pn;
            nrec[b * Rc + 1 + t * Kn + k] = pn;
        }
        __threadfence();
        __syncthreads();

        // ---- Phase 6: mk scores: dot(emb[neigh], W_mk[:256]) + hdot[b] + b_mk ----
        for (int p = wave; p < Bn * Kn; p += NWAVES) {
            int b = p / Kn;
            int nb = neighB[p];
            const float4* er = (const float4*)&embedding[(long)nb * En + lane * 8];
            const float4* wr = (const float4*)&W_mk[lane * 8];
            float4 e0 = er[0], e1 = er[1];
            float4 w0 = wr[0], w1 = wr[1];
            float s = e0.x * w0.x + e0.y * w0.y + e0.z * w0.z + e0.w * w0.w
                    + e1.x * w1.x + e1.y * w1.y + e1.z * w1.z + e1.w * w1.w;
#pragma unroll
            for (int off = 16; off > 0; off >>= 1) s += __shfl_down(s, off, 32);
            if (lane == 0) score[p] = s + hdot[b] + b_mk[0];
        }
        __threadfence();
        __syncthreads();

        // ---- Phase 7: softmax + bookkeeping + categorical sample (one thread per b) ----
        if (tid < Bn) {
            const int b = tid;
            float sc[Kn];
            float mx = -1e30f;
#pragma unroll
            for (int k = 0; k < Kn; ++k) { sc[k] = score[b * Kn + k]; mx = fmaxf(mx, sc[k]); }
            float sum = 0.0f;
#pragma unroll
            for (int k = 0; k < Kn; ++k) { sc[k] = expf(sc[k] - mx); sum += sc[k]; }
            const float inv = 1.0f / sum;

            const int   ch       = chosenA[b];
            const float chosen_p = prob[b * Lc + ch];

            prob[b * Lc + ch] = chosen_p * sc[0] * inv;
#pragma unroll
            for (int k = 1; k < Kn; ++k) {
                prob[b * Lc + 1 + t * (Kn - 1) + (k - 1)] = chosen_p * sc[k] * inv;
                cand[b * Lc + 1 + t * (Kn - 1) + (k - 1)] = neighB[b * Kn + k];
            }

            const int active = 1 + (t + 1) * (Kn - 1);
            float tot = 0.0f;
            for (int i = 0; i < active; ++i) tot += prob[b * Lc + i];
            float u = rng01((unsigned)t, (unsigned)b) * tot;
            int pick = active - 1;
            float csum = 0.0f;
            for (int i = 0; i < active; ++i) {
                csum += prob[b * Lc + i];
                if (u < csum) { pick = i; break; }
            }

            const int   nm  = cand[b * Lc + pick];
            const float snp = nrec[b * Rc + pick];
            const float ssp = prob[b * Lc + pick];
            const float nt  = newT[b];

            out[0 * BS + b * Sn + (t + 1)] = (float)nm;
            out[1 * BS + b * Sn + (t + 1)] = nt;
            out[2 * BS + b * Sn + (t + 1)] = (nt < MAX_TIMEc) ? 1.0f : 0.0f;
            out[3 * BS + b * Sn + (t + 1)] = snp;
            out[4 * BS + b * Sn + (t + 1)] = ssp;

            lastM[b]   = nm;
            lastT[b]   = nt;
            chosenA[b] = pick;
        }
        __threadfence();
        __syncthreads();
    }
}

// ---------------- host launcher ----------------
extern "C" void kernel_launch(void* const* d_in, const int* in_sizes, int n_in,
                              void* d_out, int out_size, void* d_ws, size_t ws_size,
                              hipStream_t stream) {
    const int*   marker        = (const int*)  d_in[0];
    const float* timed         = (const float*)d_in[1];
    const float* maskd         = (const float*)d_in[2];
    const float* embedding     = (const float*)d_in[3];
    const int*   neighbor_list = (const int*)  d_in[4];
    const float* neighbor_prob = (const float*)d_in[5];
    const float* W_te   = (const float*)d_in[6];
    const float* b_te   = (const float*)d_in[7];
    const float* W_el   = (const float*)d_in[8];
    const float* b_el   = (const float*)d_in[9];
    const float* W_ih   = (const float*)d_in[10];
    const float* b_ih   = (const float*)d_in[11];
    const float* W_hh   = (const float*)d_in[12];
    const float* b_hh   = (const float*)d_in[13];
    const float* W_time = (const float*)d_in[14];
    const float* b_time = (const float*)d_in[15];
    const float* W_mk   = (const float*)d_in[16];
    const float* b_mk   = (const float*)d_in[17];

    // carve workspace (256B aligned slices)
    char* p = (char*)d_ws;
    auto carve = [&](size_t bytes) -> char* {
        char* r = p;
        p += (bytes + 255) & ~size_t(255);
        return r;
    };
    __bf16* bWelT   = (__bf16*)carve((size_t)En * Dn * sizeof(__bf16));
    __bf16* bWihT   = (__bf16*)carve((size_t)Dn * Dn * sizeof(__bf16));
    __bf16* bWhhT   = (__bf16*)carve((size_t)Dn * Dn * sizeof(__bf16));
    int*    cand    = (int*)   carve((size_t)Bn * Lc * sizeof(int));
    float*  prob    = (float*) carve((size_t)Bn * Lc * sizeof(float));
    float*  nrec    = (float*) carve((size_t)Bn * Rc * sizeof(float));
    float*  lastT   = (float*) carve(Bn * sizeof(float));
    int*    lastM   = (int*)   carve(Bn * sizeof(int));
    int*    chosenA = (int*)   carve(Bn * sizeof(int));
    float*  newT    = (float*) carve(Bn * sizeof(float));
    float*  hdot    = (float*) carve(Bn * sizeof(float));
    float*  score   = (float*) carve((size_t)Bn * Kn * sizeof(float));
    int*    neighB  = (int*)   carve((size_t)Bn * Kn * sizeof(int));
    float*  pneighB = (float*) carve((size_t)Bn * Kn * sizeof(float));

    // 1) bf16 transposed weight images (L2-resident working set ~1.3 MB)
    {
        int n = Dn * Dn;
        prep_weights<<<(n + 255) / 256, 256, 0, stream>>>(W_el, W_ih, W_hh,
                                                          bWelT, bWihT, bWhhT);
    }

    // 2) persistent single-WGP scan: 32 waves, 320 KB LDS (vec/x/h in bf16)
    size_t shmem = (size_t)(Bn * En + 2 * Bn * Dn) * sizeof(__bf16);
    rnn_generator<<<1, NTHREADS, shmem, stream>>>(
        marker, timed, maskd, embedding, neighbor_list, neighbor_prob,
        W_te, b_te, bWelT, b_el, bWihT, b_ih, bWhhT, b_hh, W_time, b_time, W_mk, b_mk,
        (float*)d_out,
        cand, prob, nrec, lastT, lastM, chosenA, newT, hdot, score, neighB, pneighB);
}
